// RSSMCore_35519379538347
// MI455X (gfx1250) — matrix operand order repaired and checked by reading, not compiled, not run on hardware
//
#include <hip/hip_runtime.h>
#include <hip/hip_bf16.h>
#include <stdint.h>

// ---------------------------------------------------------------------------
// RSSM core for MI455X (gfx1250, wave32, WMMA + TDM).
// f16 weights, transposed once per launch (N x K layout); f32 accumulation via
// v_wmma_f32_16x16x32_f16. Double-buffered LDS, K-step 64 (8 WMMA / barrier).
// Weight tiles are DMA'd into LDS by the Tensor Data Mover (tensor_load_to_lds
// with LDS padding reproducing the 72-half row stride); activations staged by
// VALU (f32 -> f16 conversion). Weights (~48MB f16) stay L2-resident across
// the 50-step scan.
// ---------------------------------------------------------------------------

typedef __attribute__((ext_vector_type(16))) _Float16 v16h;
typedef __attribute__((ext_vector_type(8)))  _Float16 v8h;
typedef __attribute__((ext_vector_type(8)))  float    v8f;
typedef __attribute__((ext_vector_type(4)))  unsigned int v4u;
typedef __attribute__((ext_vector_type(8)))  int      v8i_t;
typedef __attribute__((ext_vector_type(4)))  int      v4i_t;

namespace rssm {
constexpr int N      = 50;
constexpr int B      = 256;
constexpr int EMBED  = 1536;
constexpr int ACT    = 64;
constexpr int DETER  = 2048;
constexpr int STOCH  = 64;
constexpr int HID    = 1024;
constexpr int G3     = 3 * DETER;       // 6144
constexpr int OUTW   = 2368;            // 128 prior + 128 post + 2048 h + 64 z
constexpr int OFF_PRIOR = 0;
constexpr int OFF_POST  = 128;
constexpr int OFF_H     = 256;
constexpr int OFF_Z     = 2304;
constexpr int LDSS   = 72;              // LDS row stride in halves (144B, 16B-aligned)
}

// ---------------------------------------------------------------------------
// TDM: DMA a [128 rows x 64 f16] tile from a row-major [rows x ld] f16 tensor
// into LDS with 16B padding after every 128B row (-> 72-half LDS row stride).
// D# per CDNA5 ISA ch.8: group0 = {flags, lds_addr, global_addr, type=2},
// group1 = {data_size=2B, pad 32DW/4DW, dims, strides}, groups 2/3 = 0 (2D).
// Must be executed by a single wave (TDM ignores EXEC; one issue per wave).
// ---------------------------------------------------------------------------
__device__ __forceinline__ void tdm_load_tile_f16(const _Float16* g,
                                                  uint32_t lds_off,
                                                  int k_elems, int ld) {
  uint64_t ga = (uint64_t)(uintptr_t)g;
  v4u g0;
  g0.x = 1u;                                     // count=1 (user descriptor)
  g0.y = lds_off;                                // lds_addr (bytes)
  g0.z = (uint32_t)ga;                           // global_addr[31:0]
  g0.w = (uint32_t)((ga >> 32) & 0x01FFFFFFull)  // global_addr[56:32]
       | (2u << 30);                             // type = 2 ("image")
  v8i_t g1;
  g1[0] = (int)((1u << 16)                       // data_size = 2 bytes
              | (1u << 20)                       // pad_enable
              | (4u << 22)                       // pad_interval: 32 DWORDs (128B)
              | (3u << 25));                     // pad_amount:   4 DWORDs (16B)
  g1[1] = (int)(((uint32_t)k_elems & 0xFFFFu) << 16);          // tensor_dim0 lo
  g1[2] = (int)((((uint32_t)k_elems >> 16) & 0xFFFFu)          // tensor_dim0 hi
              | (128u << 16));                                  // tensor_dim1 = 128
  g1[3] = (int)(64u << 16);                      // tensor_dim1 hi=0, tile_dim0 = 64
  g1[4] = (int)128u;                             // tile_dim1 = 128, tile_dim2 = 0
  g1[5] = (int)(uint32_t)ld;                     // tensor_dim0_stride lo (elements)
  g1[6] = 0;                                     // stride0 hi, stride1 lo
  g1[7] = 0;                                     // stride1 hi
  v4i_t gz = {0, 0, 0, 0};
#if __clang_major__ >= 23
  v8i_t gz8 = {0, 0, 0, 0, 0, 0, 0, 0};
  __builtin_amdgcn_tensor_load_to_lds(g0, g1, gz, gz, gz8, 0);
#else
  __builtin_amdgcn_tensor_load_to_lds(g0, g1, gz, gz, 0);
#endif
}

// ---------------------------------------------------------------------------
// f32 -> f16 convert + transpose: src[K x N] row-major -> dst[N x K] row-major
// ---------------------------------------------------------------------------
__global__ __launch_bounds__(256)
void cvt_transpose_kernel(const float* __restrict__ src,
                          _Float16* __restrict__ dst, int K, int Nn) {
  __shared__ _Float16 tile[32][33];
  const int nt = blockIdx.x * 32;
  const int kt = blockIdx.y * 32;
  const int tx = threadIdx.x & 31;
  const int ty = threadIdx.x >> 5;     // 0..7
#pragma unroll
  for (int i = 0; i < 32; i += 8)
    tile[ty + i][tx] = (_Float16)src[(long)(kt + ty + i) * Nn + nt + tx];
  __syncthreads();
#pragma unroll
  for (int i = 0; i < 32; i += 8)
    dst[(long)(nt + ty + i) * K + kt + tx] = tile[tx][ty + i];
}

// ---------------------------------------------------------------------------
// WMMA GEMM: ACC=0: C = bias + A*B    ACC=1: C += A*B
//   A: f32 [M x K] row-major (lda), BT: f16 [N x K] row-major (ldbt = K)
// Block = 256 threads (8 waves), tile 64(M) x 128(N), K-step 64,
// double-buffered LDS; B tiles via TDM, A tiles via VALU convert.
// ---------------------------------------------------------------------------
template <bool ACC>
__global__ __launch_bounds__(256)
void wmma_gemm_kernel(const float* __restrict__ A, int lda,
                      const _Float16* __restrict__ BT, int ldbt,
                      float* __restrict__ C, int ldc, int K,
                      const float* __restrict__ bias) {
  using namespace rssm;
  __shared__ _Float16 Al[2][64 * LDSS];     // A tile  [64 rows x 64 k-halves]
  __shared__ _Float16 Bl[2][128 * LDSS];    // B tile  [128 cols x 64 k-halves]

  const int tid  = threadIdx.x;
  const int lane = tid & 31;
  const int wave = tid >> 5;
  const int wm   = wave & 1;
  const int wn   = wave >> 1;
  const long m0  = (long)blockIdx.y * 64;
  const long n0  = (long)blockIdx.x * 128;

  v8f acc00 = {}, acc01 = {}, acc10 = {}, acc11 = {};

  // staging coordinates
  const int arow  = tid >> 2;               // 0..63
  const int acolf = (tid & 3) * 16;         // 0,16,32,48

  // fragment coordinates
  const int fr  = lane & 15;
  const int akb = (lane >> 4) << 3;         // A: K base 0/8 within 32-k step
  const int bkb = (lane >> 4) << 3;         // B: K base 0/8 within 32-k step

  const _Float16* Bbase = BT + n0 * (long)ldbt;
  const uint32_t lds_b0 = (uint32_t)(uintptr_t)(&Bl[0][0]);
  const uint32_t lds_b1 = (uint32_t)(uintptr_t)(&Bl[1][0]);

  auto fetchA = [&](int k0, float* r) {
    const float4* a4 = (const float4*)(A + (m0 + arow) * (long)lda + k0 + acolf);
#pragma unroll
    for (int i = 0; i < 4; ++i) {
      float4 v = a4[i];
      r[4 * i + 0] = v.x; r[4 * i + 1] = v.y; r[4 * i + 2] = v.z; r[4 * i + 3] = v.w;
    }
  };
  auto putA = [&](int buf, const float* r) {
    v8h lo, hi;
#pragma unroll
    for (int i = 0; i < 8; ++i) { lo[i] = (_Float16)r[i]; hi[i] = (_Float16)r[8 + i]; }
    *(v8h*)(&Al[buf][arow * LDSS + acolf])     = lo;
    *(v8h*)(&Al[buf][arow * LDSS + acolf + 8]) = hi;
  };

  auto loadA = [&](int buf, int rowbase, int s) -> v16h {
    const _Float16* p = &Al[buf][(rowbase + fr) * LDSS + s + akb];
    v8h lo = *(const v8h*)p;
    v8h hi = *(const v8h*)(p + 16);
    v16h f;
#pragma unroll
    for (int i = 0; i < 8; ++i) { f[i] = lo[i]; f[i + 8] = hi[i]; }
    return f;
  };
  auto loadB = [&](int buf, int colbase, int s) -> v16h {
    const _Float16* p = &Bl[buf][(colbase + fr) * LDSS + s + bkb];
    v8h lo = *(const v8h*)p;
    v8h hi = *(const v8h*)(p + 16);
    v16h f;
#pragma unroll
    for (int i = 0; i < 8; ++i) { f[i] = lo[i]; f[i + 8] = hi[i]; }
    return f;
  };

  // prologue: stage tile 0 (B via TDM, A via VALU convert)
  {
    float ar[16];
    if (wave == 0) tdm_load_tile_f16(Bbase, lds_b0, K, ldbt);
    fetchA(0, ar);
    putA(0, ar);
    if (wave == 0) __builtin_amdgcn_s_wait_tensorcnt(0);
  }
  __syncthreads();

  const int nk = K >> 6;
  for (int it = 0; it < nk; ++it) {
    const int cur  = it & 1;
    const bool more = (it + 1 < nk);
    float ar[16];
    if (more) {
      const int kn = (it + 1) << 6;
      if (wave == 0)
        tdm_load_tile_f16(Bbase + kn, cur ? lds_b0 : lds_b1, K, ldbt);
      fetchA(kn, ar);
      // keep the weight stream warm in L2 (global_prefetch_b8)
      __builtin_prefetch(Bbase + (long)(tid >> 1) * ldbt + kn + 64, 0, 3);
    }
    // 2 sub-steps x 4 WMMA on current buffer
#pragma unroll
    for (int s = 0; s < 64; s += 32) {
      v16h a0 = loadA(cur, wm * 32,      s);
      v16h a1 = loadA(cur, wm * 32 + 16, s);
      v16h b0 = loadB(cur, wn * 32,      s);
      v16h b1 = loadB(cur, wn * 32 + 16, s);
      acc00 = __builtin_amdgcn_wmma_f32_16x16x32_f16(false, a0, false, b0, (short)0, acc00, false, false);
      acc01 = __builtin_amdgcn_wmma_f32_16x16x32_f16(false, a0, false, b1, (short)0, acc01, false, false);
      acc10 = __builtin_amdgcn_wmma_f32_16x16x32_f16(false, a1, false, b0, (short)0, acc10, false, false);
      acc11 = __builtin_amdgcn_wmma_f32_16x16x32_f16(false, a1, false, b1, (short)0, acc11, false, false);
    }
    if (more) {
      putA(cur ^ 1, ar);
      if (wave == 0) __builtin_amdgcn_s_wait_tensorcnt(0);
    }
    __syncthreads();
  }

  // epilogue: lane L -> row = base + (L>=16?8:0) + v, col = base + L%16
  const int crow = (lane >> 4) << 3;
  const int ccol = lane & 15;
  auto storeT = [&](const v8f& a, int tm, int tn) {
    long row = m0 + wm * 32 + tm + crow;
    long col = n0 + wn * 32 + tn + ccol;
    float* cp = C + row * (long)ldc + col;
    if constexpr (ACC) {
#pragma unroll
      for (int v = 0; v < 8; ++v) cp[(long)v * ldc] += a[v];
    } else {
      float bv = bias[col];
#pragma unroll
      for (int v = 0; v < 8; ++v) cp[(long)v * ldc] = a[v] + bv;
    }
  };
  storeT(acc00, 0, 0);
  storeT(acc01, 0, 16);
  storeT(acc10, 16, 0);
  storeT(acc11, 16, 16);
}

// ---------------------------------------------------------------------------
// LayerNorm + ELU over rows of width HID=1024; 1 block (128 thr) per row
// ---------------------------------------------------------------------------
__global__ __launch_bounds__(128)
void ln_elu_kernel(const float* __restrict__ x, const float* __restrict__ g,
                   const float* __restrict__ b, float* __restrict__ y) {
  using namespace rssm;
  __shared__ float red[128];
  const long row = blockIdx.x;
  const float* xr = x + row * HID;
  float lv[8];
  float s = 0.f;
#pragma unroll
  for (int i = 0; i < 8; ++i) { lv[i] = xr[threadIdx.x + i * 128]; s += lv[i]; }
  red[threadIdx.x] = s;
  __syncthreads();
  for (int st = 64; st > 0; st >>= 1) {
    if (threadIdx.x < st) red[threadIdx.x] += red[threadIdx.x + st];
    __syncthreads();
  }
  float mean = red[0] * (1.0f / HID);
  __syncthreads();
  float vs = 0.f;
#pragma unroll
  for (int i = 0; i < 8; ++i) { float d = lv[i] - mean; vs += d * d; }
  red[threadIdx.x] = vs;
  __syncthreads();
  for (int st = 64; st > 0; st >>= 1) {
    if (threadIdx.x < st) red[threadIdx.x] += red[threadIdx.x + st];
    __syncthreads();
  }
  float rstd = rsqrtf(red[0] * (1.0f / HID) + 1e-3f);
  float* yr = y + row * HID;
#pragma unroll
  for (int i = 0; i < 8; ++i) {
    int c = threadIdx.x + i * 128;
    float v = (lv[i] - mean) * rstd * g[c] + b[c];
    yr[c] = v > 0.f ? v : (__expf(v) - 1.f);
  }
}

// ---------------------------------------------------------------------------
// reset mask: h *= m, z *= m   (idx over B*DETER; z handled for d < STOCH)
// ---------------------------------------------------------------------------
__global__ void mask_kernel(float* __restrict__ h, float* __restrict__ z,
                            const uint8_t* __restrict__ reset_t) {
  using namespace rssm;
  int idx = blockIdx.x * blockDim.x + threadIdx.x;
  int b = idx >> 11;
  int d = idx & (DETER - 1);
  float m = reset_t[b] ? 0.f : 1.f;
  h[idx] *= m;
  if (d < STOCH) z[b * STOCH + d] *= m;
}

__device__ __forceinline__ float sigmoidf_(float x) {
  return 1.0f / (1.0f + __expf(-x));
}

// ---------------------------------------------------------------------------
// GRU combine: h = (1-u)*c + u*h ; also writes h into output features slot
// ---------------------------------------------------------------------------
__global__ void gru_kernel(const float* __restrict__ gi,
                           const float* __restrict__ gh,
                           float* __restrict__ h,
                           float* __restrict__ out_h) {
  using namespace rssm;
  int idx = blockIdx.x * blockDim.x + threadIdx.x;   // < B*DETER
  int b = idx >> 11;
  int d = idx & (DETER - 1);
  const float* gib = gi + (long)b * G3;
  const float* ghb = gh + (long)b * G3;
  float r = sigmoidf_(gib[d]             + ghb[d]);
  float u = sigmoidf_(gib[DETER + d]     + ghb[DETER + d]);
  float c = tanhf(    gib[2 * DETER + d] + r * ghb[2 * DETER + d]);
  float hn = (1.f - u) * c + u * h[idx];
  h[idx] = hn;
  out_h[(long)b * OUTW + d] = hn;
}

// ---------------------------------------------------------------------------
// posterior write + reparameterized sample; writes posts & z into output
// ---------------------------------------------------------------------------
__global__ void post_kernel(const float* __restrict__ post,
                            const float* __restrict__ noise_t,
                            float* __restrict__ z,
                            float* __restrict__ out_t) {
  using namespace rssm;
  int idx = blockIdx.x * blockDim.x + threadIdx.x;   // < B*128
  int b = idx >> 7;
  int j = idx & 127;
  float p = post[idx];
  out_t[(long)b * OUTW + OFF_POST + j] = p;
  if (j < STOCH) {
    float mean = post[b * 128 + j];
    float raw  = post[b * 128 + STOCH + j];
    float std  = 2.0f * sigmoidf_(raw) + 0.1f;
    float zv = mean + std * noise_t[b * STOCH + j];
    z[b * STOCH + j] = zv;
    out_t[(long)b * OUTW + OFF_Z + j] = zv;
  }
}

// ---------------------------------------------------------------------------
// host side
// ---------------------------------------------------------------------------
static inline void launch_gemm(hipStream_t s, const float* A, int lda,
                               const _Float16* BT, int ldbt, float* C, int ldc,
                               int M, int Nn, int K, const float* bias, bool acc) {
  dim3 grid(Nn / 128, M / 64);
  if (acc)
    wmma_gemm_kernel<true><<<grid, 256, 0, s>>>(A, lda, BT, ldbt, C, ldc, K, nullptr);
  else
    wmma_gemm_kernel<false><<<grid, 256, 0, s>>>(A, lda, BT, ldbt, C, ldc, K, bias);
}

static inline void launch_cvt_t(hipStream_t s, const float* src, _Float16* dst,
                                int K, int Nn) {
  dim3 grid(Nn / 32, K / 32);
  cvt_transpose_kernel<<<grid, 256, 0, s>>>(src, dst, K, Nn);
}

extern "C" void kernel_launch(void* const* d_in, const int* in_sizes, int n_in,
                              void* d_out, int out_size, void* d_ws, size_t ws_size,
                              hipStream_t stream) {
  using namespace rssm;
  const float*   embed  = (const float*)d_in[0];
  const float*   action = (const float*)d_in[1];
  const uint8_t* reset  = (const uint8_t*)d_in[2];
  const float*   h0     = (const float*)d_in[3];
  const float*   z0     = (const float*)d_in[4];
  const float*   noises = (const float*)d_in[5];
  const float*   z_w    = (const float*)d_in[6];
  const float*   z_b    = (const float*)d_in[7];
  const float*   a_w    = (const float*)d_in[8];
  const float*   in_g   = (const float*)d_in[9];
  const float*   in_b   = (const float*)d_in[10];
  const float*   gru_wi = (const float*)d_in[11];
  const float*   gru_wh = (const float*)d_in[12];
  const float*   gru_bi = (const float*)d_in[13];
  const float*   gru_bh = (const float*)d_in[14];
  const float*   ph_w   = (const float*)d_in[15];
  const float*   ph_b   = (const float*)d_in[16];
  const float*   pe_w   = (const float*)d_in[17];
  const float*   pon_g  = (const float*)d_in[18];
  const float*   pon_b  = (const float*)d_in[19];
  const float*   pom_w  = (const float*)d_in[20];
  const float*   pom_b  = (const float*)d_in[21];
  const float*   prh_w  = (const float*)d_in[22];
  const float*   prh_b  = (const float*)d_in[23];
  const float*   prn_g  = (const float*)d_in[24];
  const float*   prn_b  = (const float*)d_in[25];
  const float*   prm_w  = (const float*)d_in[26];
  const float*   prm_b  = (const float*)d_in[27];
  float* out = (float*)d_out;

  // ---- workspace carve ----
  char* wp = (char*)d_ws;
  auto carve = [&](size_t bytes) -> void* {
    void* p = (void*)wp;
    wp += (bytes + 255) & ~(size_t)255;
    return p;
  };
  // transposed f16 weights: [N x K]
  _Float16* w_zw  = (_Float16*)carve((size_t)STOCH * HID * 2);
  _Float16* w_aw  = (_Float16*)carve((size_t)ACT * HID * 2);
  _Float16* w_wi  = (_Float16*)carve((size_t)HID * G3 * 2);
  _Float16* w_wh  = (_Float16*)carve((size_t)DETER * G3 * 2);
  _Float16* w_ph  = (_Float16*)carve((size_t)DETER * HID * 2);
  _Float16* w_pe  = (_Float16*)carve((size_t)EMBED * HID * 2);
  _Float16* w_pom = (_Float16*)carve((size_t)HID * 128 * 2);
  _Float16* w_prh = (_Float16*)carve((size_t)DETER * HID * 2);
  _Float16* w_prm = (_Float16*)carve((size_t)HID * 128 * 2);
  float* xb    = (float*)carve((size_t)B * HID * 4);
  float* za    = (float*)carve((size_t)B * HID * 4);
  float* pin   = (float*)carve((size_t)B * HID * 4);
  float* gi    = (float*)carve((size_t)B * G3 * 4);
  float* gh    = (float*)carve((size_t)B * G3 * 4);
  float* hbuf  = (float*)carve((size_t)B * DETER * 4);
  float* zbuf  = (float*)carve((size_t)B * STOCH * 4);
  float* postb = (float*)carve((size_t)B * 128 * 4);

  // ---- convert + transpose weights to f16 [N x K] (once per launch) ----
  launch_cvt_t(stream, z_w,    w_zw,  STOCH, HID);
  launch_cvt_t(stream, a_w,    w_aw,  ACT,   HID);
  launch_cvt_t(stream, gru_wi, w_wi,  HID,   G3);
  launch_cvt_t(stream, gru_wh, w_wh,  DETER, G3);
  launch_cvt_t(stream, ph_w,   w_ph,  DETER, HID);
  launch_cvt_t(stream, pe_w,   w_pe,  EMBED, HID);
  launch_cvt_t(stream, pom_w,  w_pom, HID,   128);
  launch_cvt_t(stream, prh_w,  w_prh, DETER, HID);
  launch_cvt_t(stream, prm_w,  w_prm, HID,   128);

  // ---- init carry ----
  hipMemcpyAsync(hbuf, h0, (size_t)B * DETER * 4, hipMemcpyDeviceToDevice, stream);
  hipMemcpyAsync(zbuf, z0, (size_t)B * STOCH * 4, hipMemcpyDeviceToDevice, stream);

  const int HD_BLOCKS = (B * DETER) / 256;   // 2048

  // ---- scan over timesteps ----
  for (int t = 0; t < N; ++t) {
    const float* e_t  = embed  + (size_t)t * B * EMBED;
    const float* a_t  = action + (size_t)t * B * ACT;
    const float* nz_t = noises + (size_t)t * B * STOCH;
    const uint8_t* r_t = reset + (size_t)t * B;
    float* out_t = out + (size_t)t * B * OUTW;

    // h *= m ; z *= m
    mask_kernel<<<HD_BLOCKS, 256, 0, stream>>>(hbuf, zbuf, r_t);

    // x = z @ z_w + z_b + a @ a_w
    launch_gemm(stream, zbuf, STOCH, w_zw, STOCH, xb, HID, B, HID, STOCH, z_b, false);
    launch_gemm(stream, a_t, ACT, w_aw, ACT, xb, HID, B, HID, ACT, nullptr, true);
    // za = elu(LN(x))
    ln_elu_kernel<<<B, 128, 0, stream>>>(xb, in_g, in_b, za);

    // gi = za @ gru_wi + bi ; gh = h @ gru_wh + bh
    launch_gemm(stream, za, HID, w_wi, HID, gi, G3, B, G3, HID, gru_bi, false);
    launch_gemm(stream, hbuf, DETER, w_wh, DETER, gh, G3, B, G3, DETER, gru_bh, false);
    // GRU combine -> h ; writes h into output features
    gru_kernel<<<HD_BLOCKS, 256, 0, stream>>>(gi, gh, hbuf, out_t + OFF_H);

    // x = h @ ph_w + ph_b + e @ pe_w
    launch_gemm(stream, hbuf, DETER, w_ph, DETER, xb, HID, B, HID, DETER, ph_b, false);
    launch_gemm(stream, e_t, EMBED, w_pe, EMBED, xb, HID, B, HID, EMBED, nullptr, true);
    // post_in = elu(LN(x))
    ln_elu_kernel<<<B, 128, 0, stream>>>(xb, pon_g, pon_b, pin);
    // post = post_in @ pom_w + pom_b
    launch_gemm(stream, pin, HID, w_pom, HID, postb, 128, B, 128, HID, pom_b, false);
    // sample z, write posts + z into output
    post_kernel<<<(B * 128) / 256, 256, 0, stream>>>(postb, nz_t, zbuf, out_t);
  }

  // ---- prior pass: priors = elu(LN(hs @ prh_w + prh_b)) @ prm_w + prm_b ----
  for (int t = 0; t < N; ++t) {
    float* out_t = out + (size_t)t * B * OUTW;
    // hs lives strided inside d_out at offset OFF_H with row stride OUTW
    launch_gemm(stream, out_t + OFF_H, OUTW, w_prh, DETER, xb, HID, B, HID, DETER, prh_b, false);
    ln_elu_kernel<<<B, 128, 0, stream>>>(xb, prn_g, prn_b, pin);
    // write priors directly into d_out cols [0,128) with ldc = OUTW
    launch_gemm(stream, pin, HID, w_prm, HID, out_t + OFF_PRIOR, OUTW, B, 128, HID, prm_b, false);
  }
}
